// ResGATBlock_75771813036522
// MI455X (gfx1250) — matrix operand order repaired and checked by reading, not compiled, not run on hardware
//
#include <hip/hip_runtime.h>
#include <hip/hip_bf16.h>

#define NN    50000
#define EE    800000
#define ETOT  850000     // EE + NN self loops
#define HIDC  128
#define NEGS  0.2f
#define BNEPS 1e-5f
#define ISQ2  0.7071067811865476f

typedef __attribute__((ext_vector_type(16))) __bf16 v16bf;
typedef __attribute__((ext_vector_type(8)))  __bf16 v8bf;
typedef __attribute__((ext_vector_type(8)))  float  v8f;

// ---------------------------------------------------------------- utilities
__global__ void k_fill(float* __restrict__ p, float v, int n) {
    int i = blockIdx.x * blockDim.x + threadIdx.x;
    if (i < n) p[i] = v;
}

__global__ void k_f32_to_bf16(const float* __restrict__ x, __bf16* __restrict__ y, int n) {
    int i = blockIdx.x * blockDim.x + threadIdx.x;
    if (i < n) y[i] = (__bf16)x[i];
}

// W: K x Nc (row-major, as in reference "x @ W") -> Wt: Nc x K bf16
__global__ void k_transpose_bf16(const float* __restrict__ W, __bf16* __restrict__ Wt,
                                 int K, int Nc) {
    int i = blockIdx.x * blockDim.x + threadIdx.x;
    if (i >= K * Nc) return;
    int n = i / K, k = i - n * K;
    Wt[(size_t)n * K + k] = (__bf16)W[(size_t)k * Nc + n];
}

// ---------------------------------------------------------------- WMMA GEMM
// C[M,Nc] = A[M,K](bf16,row-major) x Bt[Nc,K](bf16, pre-transposed B) (+bias)
// One wave per 16x16 output tile; K in steps of 32 via v_wmma_f32_16x16x32_bf16.
__global__ void k_gemm_wmma(const __bf16* __restrict__ A, const __bf16* __restrict__ Bt,
                            const float* __restrict__ bias, float* __restrict__ C,
                            int M, int K, int Nc) {
    int wid  = (blockIdx.x * blockDim.x + threadIdx.x) >> 5;
    int lane = threadIdx.x & 31;
    int tilesN = Nc >> 4;
    int tm = wid / tilesN;
    int tn = wid - tm * tilesN;
    if (tm * 16 >= M) return;
    int half = lane >> 4;          // 0: lanes 0-15, 1: lanes 16-31
    int r    = lane & 15;

    const __bf16* arow = A  + (size_t)(tm * 16 + r) * K;
    const __bf16* brow = Bt + (size_t)(tn * 16 + r) * K;

    v8f acc = {0.f, 0.f, 0.f, 0.f, 0.f, 0.f, 0.f, 0.f};

    for (int k0 = 0; k0 < K; k0 += 32) {
        // A fragment: lanes<16 hold K k0+0..7 & k0+16..23; lanes>=16 shifted by 8
        int ka = k0 + half * 8;
        v8bf a0 = *(const v8bf*)(arow + ka);
        v8bf a1 = *(const v8bf*)(arow + ka + 16);
        // B fragment: lanes<16 hold K k0+0..7 & k0+16..23; lanes>=16 shifted by 8
        int kb = k0 + half * 8;
        v8bf b0 = *(const v8bf*)(brow + kb);
        v8bf b1 = *(const v8bf*)(brow + kb + 16);
        v16bf av = __builtin_shufflevector(a0, a1, 0,1,2,3,4,5,6,7,8,9,10,11,12,13,14,15);
        v16bf bv = __builtin_shufflevector(b0, b1, 0,1,2,3,4,5,6,7,8,9,10,11,12,13,14,15);
        acc = __builtin_amdgcn_wmma_f32_16x16x32_bf16(false, av, false, bv,
                                                      (short)0, acc, false, false);
    }

    // C/D layout: VGPR i -> M = tm*16 + i + half*8, N = tn*16 + r
    int n = tn * 16 + r;
    float bv2 = bias ? bias[n] : 0.f;
#pragma unroll
    for (int i = 0; i < 8; ++i)
        C[(size_t)(tm * 16 + i + half * 8) * Nc + n] = acc[i] + bv2;
}

// ---------------------------------------------------------------- attention
// alpha_s[n,h] = dot(H[n, h*C:(h+1)*C], att_src[h]); same for dst
__global__ void k_alpha(const float* __restrict__ H, const float* __restrict__ a_src,
                        const float* __restrict__ a_dst, float* __restrict__ as,
                        float* __restrict__ ad, int N, int heads, int C) {
    int idx = blockIdx.x * blockDim.x + threadIdx.x;
    if (idx >= N * heads) return;
    int n = idx / heads, h = idx - n * heads;
    const float* row = H + (size_t)n * heads * C + (size_t)h * C;
    const float* ws  = a_src + (size_t)h * C;
    const float* wd  = a_dst + (size_t)h * C;
    float s = 0.f, d = 0.f;
    for (int c = 0; c < C; ++c) { s += row[c] * ws[c]; d += row[c] * wd[c]; }
    as[idx] = s; ad[idx] = d;
}

__device__ __forceinline__ void atomicMaxF(float* addr, float v) {
    if (v >= 0.f) atomicMax((int*)addr, __float_as_int(v));
    else          atomicMin((unsigned int*)addr, __float_as_uint(v));
}

__global__ void k_edge_max(const int* __restrict__ src, const int* __restrict__ dst,
                           const float* __restrict__ as, const float* __restrict__ ad,
                           float* __restrict__ mx, int heads) {
    int idx = blockIdx.x * blockDim.x + threadIdx.x;
    if (idx >= ETOT * heads) return;
    int e = idx / heads, h = idx - e * heads;
    int s = (e < EE) ? src[e] : (e - EE);
    int d = (e < EE) ? dst[e] : (e - EE);
    float v = as[s * heads + h] + ad[d * heads + h];
    v = (v > 0.f) ? v : NEGS * v;
    atomicMaxF(&mx[d * heads + h], v);
}

__global__ void k_edge_exp(const int* __restrict__ src, const int* __restrict__ dst,
                           const float* __restrict__ as, const float* __restrict__ ad,
                           const float* __restrict__ mx, float* __restrict__ den,
                           float* __restrict__ eexp, int heads) {
    int idx = blockIdx.x * blockDim.x + threadIdx.x;
    if (idx >= ETOT * heads) return;
    int e = idx / heads, h = idx - e * heads;
    int s = (e < EE) ? src[e] : (e - EE);
    int d = (e < EE) ? dst[e] : (e - EE);
    float v = as[s * heads + h] + ad[d * heads + h];
    v = (v > 0.f) ? v : NEGS * v;
    float t = __expf(v - mx[d * heads + h]);
    eexp[idx] = t;
    atomicAdd(&den[d * heads + h], t);
}

// one wave per edge; lane handles 4 consecutive channels (float4 gather + 4 atomics)
__global__ void k_scatter(const int* __restrict__ src, const int* __restrict__ dst,
                          const float* __restrict__ H, const float* __restrict__ eexp,
                          const float* __restrict__ den, float* __restrict__ agg,
                          int heads, int cshift) {
    int wid  = (blockIdx.x * blockDim.x + threadIdx.x) >> 5;
    int lane = threadIdx.x & 31;
    if (wid >= ETOT) return;
    int s = (wid < EE) ? src[wid] : (wid - EE);
    int d = (wid < EE) ? dst[wid] : (wid - EE);
    int c0 = lane * 4;
    int h  = c0 >> cshift;                          // C is 32 or 128 -> same head for 4 ch
    float coef = eexp[(size_t)wid * heads + h] / (den[d * heads + h] + 1e-16f);
    const float4 hv = *(const float4*)(H + (size_t)s * HIDC + c0);
    float* ao = agg + (size_t)d * HIDC + c0;
    atomicAdd(ao + 0, hv.x * coef);
    atomicAdd(ao + 1, hv.y * coef);
    atomicAdd(ao + 2, hv.z * coef);
    atomicAdd(ao + 3, hv.w * coef);
}

// ---------------------------------------------------------------- batch norm
__global__ void k_bn_stats(const float* __restrict__ X, float* __restrict__ mean,
                           float* __restrict__ istd, int N) {
    __shared__ float ss[256], ss2[256];
    int j = blockIdx.x;                               // one block per column
    float s = 0.f, s2 = 0.f;
    for (int i = threadIdx.x; i < N; i += 256) {
        float v = X[(size_t)i * HIDC + j];
        s += v; s2 += v * v;
    }
    ss[threadIdx.x] = s; ss2[threadIdx.x] = s2;
    __syncthreads();
    for (int off = 128; off > 0; off >>= 1) {
        if ((int)threadIdx.x < off) {
            ss[threadIdx.x]  += ss[threadIdx.x + off];
            ss2[threadIdx.x] += ss2[threadIdx.x + off];
        }
        __syncthreads();
    }
    if (threadIdx.x == 0) {
        float mu  = ss[0] / (float)N;
        float var = ss2[0] / (float)N - mu * mu;
        mean[j] = mu;
        istd[j] = rsqrtf(var + BNEPS);
    }
}

// relu(bn(x)) -> bf16 (layer-1 output feeds next WMMA GEMM)
__global__ void k_bn_apply_relu(const float* __restrict__ X, const float* __restrict__ mean,
                                const float* __restrict__ istd, const float* __restrict__ gamma,
                                const float* __restrict__ beta, __bf16* __restrict__ Yb,
                                int total) {
    int idx = blockIdx.x * blockDim.x + threadIdx.x;
    if (idx >= total) return;
    int j = idx & (HIDC - 1);
    float v = (X[idx] - mean[j]) * istd[j] * gamma[j] + beta[j];
    Yb[idx] = (__bf16)fmaxf(v, 0.f);
}

// (relu(bn(x)) + res) * 1/sqrt(2) -> bf16 (feeds final WMMA GEMM)
__global__ void k_bn_res_fuse(const float* __restrict__ X, const float* __restrict__ mean,
                              const float* __restrict__ istd, const float* __restrict__ gamma,
                              const float* __restrict__ beta, const float* __restrict__ res,
                              __bf16* __restrict__ Zb, int total) {
    int idx = blockIdx.x * blockDim.x + threadIdx.x;
    if (idx >= total) return;
    int j = idx & (HIDC - 1);
    float v = (X[idx] - mean[j]) * istd[j] * gamma[j] + beta[j];
    v = fmaxf(v, 0.f);
    Zb[idx] = (__bf16)((v + res[idx]) * ISQ2);
}

// ---------------------------------------------------------------- launch
static inline size_t al256(size_t x) { return (x + 255) & ~(size_t)255; }

extern "C" void kernel_launch(void* const* d_in, const int* in_sizes, int n_in,
                              void* d_out, int out_size, void* d_ws, size_t ws_size,
                              hipStream_t stream) {
    const float* x        = (const float*)d_in[0];
    const int*   eidx     = (const int*)  d_in[1];
    const float* W1       = (const float*)d_in[2];
    const float* att_s1   = (const float*)d_in[3];
    const float* att_d1   = (const float*)d_in[4];
    const float* W2       = (const float*)d_in[6];
    const float* att_s2   = (const float*)d_in[7];
    const float* att_d2   = (const float*)d_in[8];
    const float* gamma1   = (const float*)d_in[10];
    const float* beta1    = (const float*)d_in[11];
    const float* gamma2   = (const float*)d_in[12];
    const float* beta2    = (const float*)d_in[13];
    const float* W_res    = (const float*)d_in[14];
    const float* b_res    = (const float*)d_in[15];
    const float* W_fin    = (const float*)d_in[16];
    const float* b_fin    = (const float*)d_in[17];
    const int* src = eidx;
    const int* dst = eidx + EE;
    float* out = (float*)d_out;

    // workspace carve-out
    char* ws = (char*)d_ws; size_t off = 0;
    auto alloc = [&](size_t bytes) { void* p = ws + off; off = al256(off + bytes); return p; };
    __bf16* XB  = (__bf16*)alloc((size_t)NN * HIDC * 2);    // x in bf16
    __bf16* W1T = (__bf16*)alloc(128 * 128 * 2);
    __bf16* W2T = (__bf16*)alloc(128 * 128 * 2);
    __bf16* WRT = (__bf16*)alloc(128 * 128 * 2);
    __bf16* WFT = (__bf16*)alloc(32 * 128 * 2);
    float*  HL  = (float*)alloc((size_t)NN * HIDC * 4);     // h = A @ W (per layer)
    float*  AGG = (float*)alloc((size_t)NN * HIDC * 4);     // aggregated messages
    float*  RES = (float*)alloc((size_t)NN * HIDC * 4);     // residual branch
    __bf16* HB  = (__bf16*)alloc((size_t)NN * HIDC * 2);    // activated bf16 feed
    float*  EEX = (float*)alloc((size_t)ETOT * 4 * 4);      // per-edge exp weights
    float*  AS  = (float*)alloc((size_t)NN * 4 * 4);
    float*  AD  = (float*)alloc((size_t)NN * 4 * 4);
    float*  MX  = (float*)alloc((size_t)NN * 4 * 4);
    float*  DEN = (float*)alloc((size_t)NN * 4 * 4);
    float*  MEAN= (float*)alloc(HIDC * 4);
    float*  ISTD= (float*)alloc(HIDC * 4);
    (void)ws_size; (void)n_in; (void)in_sizes; (void)out_size;

    const int T = 256;
    const int elemsNH = NN * HIDC;                  // 6.4M
    const int gNH = (elemsNH + T - 1) / T;
    const int gemmBlocks128 = ((NN / 16) * 8 * 32 + T - 1) / T;  // 3125
    const int gemmBlocks32  = ((NN / 16) * 2 * 32 + T - 1) / T;  // 782
    const int scatBlocks    = (ETOT * 32 + T - 1) / T;           // 106250

    // --- prep: weights (transposed bf16) and x (bf16)
    k_transpose_bf16<<<(128 * 128 + T - 1) / T, T, 0, stream>>>(W1,   W1T, 128, 128);
    k_transpose_bf16<<<(128 * 128 + T - 1) / T, T, 0, stream>>>(W2,   W2T, 128, 128);
    k_transpose_bf16<<<(128 * 128 + T - 1) / T, T, 0, stream>>>(W_res,WRT, 128, 128);
    k_transpose_bf16<<<(128 * 32  + T - 1) / T, T, 0, stream>>>(W_fin,WFT, 128, 32);
    k_f32_to_bf16<<<gNH, T, 0, stream>>>(x, XB, elemsNH);

    // ================= layer 1 (heads=4, C=32) =================
    k_gemm_wmma<<<gemmBlocks128, T, 0, stream>>>(XB, W1T, nullptr, HL, NN, 128, 128);
    k_alpha<<<(NN * 4 + T - 1) / T, T, 0, stream>>>(HL, att_s1, att_d1, AS, AD, NN, 4, 32);
    k_fill<<<(NN * 4 + T - 1) / T, T, 0, stream>>>(MX, -3.0e38f, NN * 4);
    k_fill<<<(NN * 4 + T - 1) / T, T, 0, stream>>>(DEN, 0.f, NN * 4);
    k_fill<<<gNH, T, 0, stream>>>(AGG, 0.f, elemsNH);
    k_edge_max<<<(ETOT * 4 + T - 1) / T, T, 0, stream>>>(src, dst, AS, AD, MX, 4);
    k_edge_exp<<<(ETOT * 4 + T - 1) / T, T, 0, stream>>>(src, dst, AS, AD, MX, DEN, EEX, 4);
    k_scatter<<<scatBlocks, T, 0, stream>>>(src, dst, HL, EEX, DEN, AGG, 4, 5);
    k_bn_stats<<<HIDC, T, 0, stream>>>(AGG, MEAN, ISTD, NN);
    k_bn_apply_relu<<<gNH, T, 0, stream>>>(AGG, MEAN, ISTD, gamma1, beta1, HB, elemsNH);

    // ================= layer 2 (heads=1, C=128) ================
    k_gemm_wmma<<<gemmBlocks128, T, 0, stream>>>(HB, W2T, nullptr, HL, NN, 128, 128);
    k_alpha<<<(NN + T - 1) / T, T, 0, stream>>>(HL, att_s2, att_d2, AS, AD, NN, 1, 128);
    k_fill<<<(NN + T - 1) / T, T, 0, stream>>>(MX, -3.0e38f, NN);
    k_fill<<<(NN + T - 1) / T, T, 0, stream>>>(DEN, 0.f, NN);
    k_fill<<<gNH, T, 0, stream>>>(AGG, 0.f, elemsNH);
    k_edge_max<<<(ETOT + T - 1) / T, T, 0, stream>>>(src, dst, AS, AD, MX, 1);
    k_edge_exp<<<(ETOT + T - 1) / T, T, 0, stream>>>(src, dst, AS, AD, MX, DEN, EEX, 1);
    k_scatter<<<scatBlocks, T, 0, stream>>>(src, dst, HL, EEX, DEN, AGG, 1, 7);
    k_bn_stats<<<HIDC, T, 0, stream>>>(AGG, MEAN, ISTD, NN);

    // residual branch + fuse -> bf16 feed for final GEMM
    k_gemm_wmma<<<gemmBlocks128, T, 0, stream>>>(XB, WRT, b_res, RES, NN, 128, 128);
    k_bn_res_fuse<<<gNH, T, 0, stream>>>(AGG, MEAN, ISTD, gamma2, beta2, RES, HB, elemsNH);

    // final projection -> d_out [N, 32]
    k_gemm_wmma<<<gemmBlocks32, T, 0, stream>>>(HB, WFT, b_fin, out, NN, 128, 32);
}
